// InputEmbedder2_48928267436116
// MI455X (gfx1250) — matrix operand order repaired and checked by reading, not compiled, not run on hardware
//
#include <hip/hip_runtime.h>
#include <hip/hip_bf16.h>
#include <math.h>

// ---------------- problem constants ----------------
#define BB     2
#define NN     384
#define TF     22
#define POS    32
#define TIMED  32
#define TPROJ  33
#define CS     256
#define CZ     128
#define PI_F        3.14159265358979f
#define LOG_POSMAX  7.62851748f      // ln(2056)
#define LOG_TIMEMAX 9.21034037f      // ln(10000)
#define LN_EPS      1e-5f

typedef float    v2f  __attribute__((ext_vector_type(2)));
typedef float    v8f  __attribute__((ext_vector_type(8)));
typedef _Float16 v8h  __attribute__((ext_vector_type(8)));
typedef _Float16 v16h __attribute__((ext_vector_type(16)));

static __device__ __forceinline__ v8f wmma_f16(v16h a, v16h b, v8f c) {
  // D(f32 16x16) = A(f16 16x32) * B(f16 32x16) + C(f32)
  return __builtin_amdgcn_wmma_f32_16x16x32_f16(false, a, false, b, (short)0, c,
                                                false, false);
}

// split fp32 -> f16 hi + f16 lo (hi*hB products are exact in f32 accumulate)
static __device__ __forceinline__ void split16(float x, _Float16& hi,
                                               _Float16& lo) {
  hi = (_Float16)x;
  lo = (_Float16)(x - (float)hi);
}

// =====================================================================
// Kernel 0: convert an fp32 weight matrix to f16 hi/lo pair (row-major)
// =====================================================================
__global__ __launch_bounds__(256) void cvt_kernel(const float* __restrict__ src,
                                                  _Float16* __restrict__ hi,
                                                  _Float16* __restrict__ lo,
                                                  int n) {
  const int i = blockIdx.x * 256 + threadIdx.x;
  if (i < n) {
    const float x = src[i];
    const _Float16 h = (_Float16)x;
    hi[i] = h;
    lo[i] = (_Float16)(x - (float)h);
  }
}

// =====================================================================
// Kernel 1: node path (seq_emb) + precompute of edge left/right vectors
// One block per (b, n) row; 256 threads (one per CS channel).
// =====================================================================
__global__ __launch_bounds__(256) void node_kernel(
    const int* __restrict__ seq_idx, const float* __restrict__ seq_feat,
    const float* __restrict__ timestep, const float* __restrict__ frame_mask,
    const float* __restrict__ W_npos, const float* __restrict__ b_npos,
    const float* __restrict__ W_ntf,  const float* __restrict__ b_ntf,
    const float* __restrict__ W_ntime,const float* __restrict__ b_ntime,
    const float* __restrict__ W_n1,   const float* __restrict__ b_n1,
    const float* __restrict__ W_n2,   const float* __restrict__ b_n2,
    const float* __restrict__ g_n,    const float* __restrict__ be_n,
    const float* __restrict__ W_etf,  const float* __restrict__ W_etime,
    float* __restrict__ out_node, float* __restrict__ leftv,
    float* __restrict__ rightv)
{
  const int n = blockIdx.x, b = blockIdx.y, tid = threadIdx.x;
  const int row = b * NN + n;

  __shared__ float s_pe[POS];
  __shared__ float s_temb[TPROJ];
  __shared__ float s_feat[TF];
  __shared__ float s_vec[CS];
  __shared__ float s_vec2[CS];

  if (tid < POS) {
    const float idx = (float)seq_idx[row];
    const int kk = tid & 15;
    const float ang = idx * (PI_F * __expf(-((float)kk / 16.0f) * LOG_POSMAX));
    s_pe[tid] = (tid < 16) ? __sinf(ang) : __cosf(ang);
  }
  if (tid < TPROJ) {
    const float fm = frame_mask[row];
    if (tid == TIMED) {
      s_temb[TIMED] = fm;
    } else {
      const float tr = timestep[b] * fm;
      const int kk = tid & 15;
      const float fr = __expf(-(LOG_TIMEMAX / 15.0f) * (float)kk);
      const float ang = tr * fr;
      s_temb[tid] = (tid < 16) ? __sinf(ang) : __cosf(ang);
    }
  }
  if (tid < TF) s_feat[tid] = seq_feat[row * TF + tid];
  __syncthreads();

  // ---- pre-activation of the node embedder input ----
  const int c = tid;
  float pre = b_npos[c] + b_ntf[c] + b_ntime[c];
  #pragma unroll 8
  for (int k = 0; k < POS; ++k)   pre += s_pe[k]   * W_npos[c * POS + k];
  for (int k = 0; k < TF; ++k)    pre += s_feat[k] * W_ntf[c * TF + k];
  for (int k = 0; k < TPROJ; ++k) pre += s_temb[k] * W_ntime[c * TPROJ + k];
  s_vec[c] = fmaxf(pre, 0.0f);

  // ---- edge left/right vectors (threads 0..127 -> left, 128..255 -> right)
  if (tid < CZ) {
    float l = 0.0f;
    for (int t = 0; t < TF; ++t)    l += s_feat[t] * W_etf[tid * (2 * TF) + t];
    for (int p = 0; p < TPROJ; ++p) l += s_temb[p] * W_etime[tid * (2 * TPROJ) + p];
    leftv[(size_t)row * CZ + tid] = l;
  } else {
    const int cc = tid - CZ;
    float r = 0.0f;
    for (int t = 0; t < TF; ++t)    r += s_feat[t] * W_etf[cc * (2 * TF) + TF + t];
    for (int p = 0; p < TPROJ; ++p) r += s_temb[p] * W_etime[cc * (2 * TPROJ) + TPROJ + p];
    rightv[(size_t)row * CZ + cc] = r;
  }
  __syncthreads();

  // ---- node MLP layer 1 ----
  float h = b_n1[c];
  #pragma unroll 8
  for (int k = 0; k < CS; ++k) h += s_vec[k] * W_n1[c * CS + k];
  s_vec2[c] = fmaxf(h, 0.0f);
  __syncthreads();

  // ---- node MLP layer 2 ----
  float o = b_n2[c];
  #pragma unroll 8
  for (int k = 0; k < CS; ++k) o += s_vec2[k] * W_n2[c * CS + k];
  __syncthreads();
  s_vec[c] = o;
  __syncthreads();

  // ---- LayerNorm over 256 channels ----
  float s1 = 0.0f, s2 = 0.0f;
  #pragma unroll 8
  for (int k = 0; k < CS; ++k) { const float x = s_vec[k]; s1 += x; s2 += x * x; }
  const float mean = s1 * (1.0f / CS);
  const float var  = s2 * (1.0f / CS) - mean * mean;
  const float rs   = rsqrtf(var + LN_EPS);
  out_node[(size_t)row * CS + c] = (o - mean) * rs * g_n[c] + be_n[c];
}

// =====================================================================
// Kernel 2: edge path.  Split-f16 emulated-fp32 GEMMs on
// V_WMMA_F32_16X16X32_F16:  A*B ~= Ah*Bh + Ah*Bl + Al*Bh  (f32 accum).
// Block = 128 threads = 4 waves. Tile = 16 i-rows x 4 j-cols of edges.
// Wave w owns column j0+w; 16 edges x 128 channels live in 8 v8f accs.
//
// Fragment layouts (k = 16h + s, same order for A and B) are plain
// contiguous runs in row-major f16 arrays:
//   A frag (lane: m=ln, half h): halves [m][32kc+16h .. +15]
//   B frag (lane: c=n0+ln, half h): halves [c][32kc+16h .. +15]
// =====================================================================
__global__ __launch_bounds__(128) void edge_kernel(
    const int* __restrict__ seq_idx,
    const _Float16* __restrict__ Wh_epos, const _Float16* __restrict__ Wl_epos,
    const float* __restrict__ b_epos, const float* __restrict__ b_etf,
    const float* __restrict__ b_etime,
    const _Float16* __restrict__ Wh_e1, const _Float16* __restrict__ Wl_e1,
    const float* __restrict__ b_e1,
    const _Float16* __restrict__ Wh_e2, const _Float16* __restrict__ Wl_e2,
    const float* __restrict__ b_e2,
    const float* __restrict__ g_e,  const float* __restrict__ be_e,
    const float* __restrict__ leftv, const float* __restrict__ rightv,
    float* __restrict__ out_edge)
{
  __shared__ float s_left[16 * CZ];
  __shared__ float s_right[4 * CZ];
  __shared__ float s_bias0[CZ], s_b1[CZ], s_b2[CZ], s_g[CZ], s_be[CZ];
  __shared__ __align__(16) _Float16 s_scrh[4][16 * CZ];  // per-wave hi scratch
  __shared__ __align__(16) _Float16 s_scrl[4][16 * CZ];  // per-wave lo scratch

  const int b  = blockIdx.z;
  const int i0 = blockIdx.y * 16;
  const int j0 = blockIdx.x * 4;
  const int tid = threadIdx.x;
  const int w    = tid >> 5;        // wave id 0..3
  const int lane = tid & 31;
  const int h    = lane >> 4;       // lane half (0/1)
  const int ln   = lane & 15;       // lane-in-half

  // ---- cooperative staging (shared across waves) ----
  for (int k = tid; k < 16 * CZ; k += 128)
    s_left[k] = leftv[((size_t)b * NN + i0) * CZ + k];
  for (int k = tid; k < 4 * CZ; k += 128)
    s_right[k] = rightv[((size_t)b * NN + j0) * CZ + k];
  if (tid < CZ) {
    s_bias0[tid] = b_epos[tid] + b_etf[tid] + b_etime[tid];
    s_b1[tid] = b_e1[tid];
    s_b2[tid] = b_e2[tid];
    s_g[tid]  = g_e[tid];
    s_be[tid] = be_e[tid];
  }
  __syncthreads();

  const int j = j0 + w;
  // rel for A-matrix row M = ln (same for both lane halves)
  const float reli =
      (float)(seq_idx[b * NN + i0 + ln] - seq_idx[b * NN + j]);

  // ---- accumulator init: bias0 + left[i] + right[j]  (exact f32) ----
  v8f acc[8];
  #pragma unroll
  for (int nt = 0; nt < 8; ++nt) {
    const int c = nt * 16 + ln;
    const float base = s_bias0[c] + s_right[w * CZ + c];
    v8f a;
    #pragma unroll
    for (int r = 0; r < 8; ++r)           // row M = r + 8*h
      a[r] = base + s_left[(r + 8 * h) * CZ + c];
    acc[nt] = a;
  }

  // ---- GEMM1: e0 += pe(rel) @ W_epos^T   (K = 32, one f16 chunk) ----
  {
    // A-fragment elements: e<8 -> K = e+8h (sin of kk=e+8h),
    //                      e>=8 -> K = 8+e+8h (cos of kk=e-8+8h).
    v16h Ah, Al;
    #pragma unroll
    for (int e = 0; e < 8; ++e) {
      const int kk = e + 8 * h;
      const float f   = PI_F * __expf(-((float)kk / 16.0f) * LOG_POSMAX);
      const float ang = reli * f;
      const float sv = __sinf(ang);
      const float cv = __cosf(ang);
      _Float16 hh, ll;
      split16(sv, hh, ll); Ah[e] = hh;     Al[e] = ll;
      split16(cv, hh, ll); Ah[e + 8] = hh; Al[e + 8] = ll;
    }
    #pragma unroll
    for (int nt = 0; nt < 8; ++nt) {
      const int c = nt * 16 + ln;
      const v8h b0h = *(const v8h*)(Wh_epos + c * POS + 8 * h);
      const v8h b1h = *(const v8h*)(Wh_epos + c * POS + 16 + 8 * h);
      const v8h b0l = *(const v8h*)(Wl_epos + c * POS + 8 * h);
      const v8h b1l = *(const v8h*)(Wl_epos + c * POS + 16 + 8 * h);
      const v16h Bh = __builtin_shufflevector(b0h, b1h, 0, 1, 2, 3, 4, 5, 6, 7,
                                              8, 9, 10, 11, 12, 13, 14, 15);
      const v16h Bl = __builtin_shufflevector(b0l, b1l, 0, 1, 2, 3, 4, 5, 6, 7,
                                              8, 9, 10, 11, 12, 13, 14, 15);
      acc[nt] = wmma_f16(Ah, Bh, acc[nt]);
      acc[nt] = wmma_f16(Ah, Bl, acc[nt]);
      acc[nt] = wmma_f16(Al, Bh, acc[nt]);
    }
  }

  _Float16* const sh = &s_scrh[w][0];
  _Float16* const sl = &s_scrl[w][0];

  // ================= GEMM2: h = relu(e0) @ W_e1^T + b_e1 =================
  #pragma unroll
  for (int nt = 0; nt < 8; ++nt)
    #pragma unroll
    for (int r = 0; r < 8; ++r) {
      const float x = fmaxf(acc[nt][r], 0.0f);
      _Float16 hh, ll;
      split16(x, hh, ll);
      const int a = (r + 8 * h) * CZ + nt * 16 + ln;
      sh[a] = hh; sl[a] = ll;
    }
  asm volatile("s_wait_dscnt 0" ::: "memory");  // intra-wave LDS hand-off

  #pragma unroll
  for (int nt = 0; nt < 8; ++nt) {
    const float bb = s_b1[nt * 16 + ln];
    acc[nt] = (v8f){bb, bb, bb, bb, bb, bb, bb, bb};
  }
  #pragma unroll
  for (int kc = 0; kc < 4; ++kc) {
    const int abase = ln * CZ + 32 * kc + 16 * h;
    const v8h a0h = *(const v8h*)(sh + abase);
    const v8h a1h = *(const v8h*)(sh + abase + 8);
    const v8h a0l = *(const v8h*)(sl + abase);
    const v8h a1l = *(const v8h*)(sl + abase + 8);
    const v16h Ah = __builtin_shufflevector(a0h, a1h, 0, 1, 2, 3, 4, 5, 6, 7,
                                            8, 9, 10, 11, 12, 13, 14, 15);
    const v16h Al = __builtin_shufflevector(a0l, a1l, 0, 1, 2, 3, 4, 5, 6, 7,
                                            8, 9, 10, 11, 12, 13, 14, 15);
    #pragma unroll
    for (int nt = 0; nt < 8; ++nt) {
      const int boff = (nt * 16 + ln) * CZ + 32 * kc + 16 * h;
      const v16h Bh = *(const v16h*)(Wh_e1 + boff);
      const v16h Bl = *(const v16h*)(Wl_e1 + boff);
      acc[nt] = wmma_f16(Ah, Bh, acc[nt]);
      acc[nt] = wmma_f16(Ah, Bl, acc[nt]);
      acc[nt] = wmma_f16(Al, Bh, acc[nt]);
    }
  }

  // ================= GEMM3: e = relu(h) @ W_e2^T + b_e2 =================
  #pragma unroll
  for (int nt = 0; nt < 8; ++nt)
    #pragma unroll
    for (int r = 0; r < 8; ++r) {
      const float x = fmaxf(acc[nt][r], 0.0f);
      _Float16 hh, ll;
      split16(x, hh, ll);
      const int a = (r + 8 * h) * CZ + nt * 16 + ln;
      sh[a] = hh; sl[a] = ll;
    }
  asm volatile("s_wait_dscnt 0" ::: "memory");

  #pragma unroll
  for (int nt = 0; nt < 8; ++nt) {
    const float bb = s_b2[nt * 16 + ln];
    acc[nt] = (v8f){bb, bb, bb, bb, bb, bb, bb, bb};
  }
  #pragma unroll
  for (int kc = 0; kc < 4; ++kc) {
    const int abase = ln * CZ + 32 * kc + 16 * h;
    const v8h a0h = *(const v8h*)(sh + abase);
    const v8h a1h = *(const v8h*)(sh + abase + 8);
    const v8h a0l = *(const v8h*)(sl + abase);
    const v8h a1l = *(const v8h*)(sl + abase + 8);
    const v16h Ah = __builtin_shufflevector(a0h, a1h, 0, 1, 2, 3, 4, 5, 6, 7,
                                            8, 9, 10, 11, 12, 13, 14, 15);
    const v16h Al = __builtin_shufflevector(a0l, a1l, 0, 1, 2, 3, 4, 5, 6, 7,
                                            8, 9, 10, 11, 12, 13, 14, 15);
    #pragma unroll
    for (int nt = 0; nt < 8; ++nt) {
      const int boff = (nt * 16 + ln) * CZ + 32 * kc + 16 * h;
      const v16h Bh = *(const v16h*)(Wh_e2 + boff);
      const v16h Bl = *(const v16h*)(Wl_e2 + boff);
      acc[nt] = wmma_f16(Ah, Bh, acc[nt]);
      acc[nt] = wmma_f16(Ah, Bl, acc[nt]);
      acc[nt] = wmma_f16(Al, Bh, acc[nt]);
    }
  }

  // ---- LayerNorm over 128 channels (from registers) + store ----
  #pragma unroll
  for (int r = 0; r < 8; ++r) {
    float s1 = 0.0f, s2 = 0.0f;
    #pragma unroll
    for (int nt = 0; nt < 8; ++nt) {
      const float x = acc[nt][r];
      s1 += x; s2 += x * x;
    }
    // reduce within each 16-lane half (rows r and r+8 independently)
    #pragma unroll
    for (int m = 1; m <= 8; m <<= 1) {
      s1 += __shfl_xor(s1, m, 32);
      s2 += __shfl_xor(s2, m, 32);
    }
    const float mean = s1 * (1.0f / CZ);
    const float var  = s2 * (1.0f / CZ) - mean * mean;
    const float rs   = rsqrtf(var + LN_EPS);
    const int mrow = r + 8 * h;
    const size_t base = (((size_t)b * NN + (i0 + mrow)) * NN + j) * CZ;
    #pragma unroll
    for (int nt = 0; nt < 8; ++nt) {
      const int c = nt * 16 + ln;
      out_edge[base + c] = (acc[nt][r] - mean) * rs * s_g[c] + s_be[c];
    }
  }
}

// =====================================================================
extern "C" void kernel_launch(void* const* d_in, const int* in_sizes, int n_in,
                              void* d_out, int out_size, void* d_ws,
                              size_t ws_size, hipStream_t stream) {
  (void)in_sizes; (void)n_in; (void)out_size; (void)ws_size;
  const int*   seq_idx    = (const int*)  d_in[0];
  const float* seq_feat   = (const float*)d_in[1];
  const float* timestep   = (const float*)d_in[2];
  const float* frame_mask = (const float*)d_in[3];
  const float* W_npos = (const float*)d_in[4],  *b_npos = (const float*)d_in[5];
  const float* W_ntf  = (const float*)d_in[6],  *b_ntf  = (const float*)d_in[7];
  const float* W_ntime= (const float*)d_in[8],  *b_ntime= (const float*)d_in[9];
  const float* W_n1   = (const float*)d_in[10], *b_n1   = (const float*)d_in[11];
  const float* W_n2   = (const float*)d_in[12], *b_n2   = (const float*)d_in[13];
  const float* g_n    = (const float*)d_in[14], *be_n   = (const float*)d_in[15];
  const float* W_epos = (const float*)d_in[16], *b_epos = (const float*)d_in[17];
  const float* W_etf  = (const float*)d_in[18], *b_etf  = (const float*)d_in[19];
  const float* W_etime= (const float*)d_in[20], *b_etime= (const float*)d_in[21];
  const float* W_e1   = (const float*)d_in[22], *b_e1   = (const float*)d_in[23];
  const float* W_e2   = (const float*)d_in[24], *b_e2   = (const float*)d_in[25];
  const float* g_e    = (const float*)d_in[26], *be_e   = (const float*)d_in[27];

  float* out_node = (float*)d_out;                       // [B,N,CS]
  float* out_edge = out_node + (size_t)BB * NN * CS;     // [B,N,N,CZ]

  // ---- workspace layout ----
  float* leftv  = (float*)d_ws;                          // [B,N,CZ] f32
  float* rightv = leftv + (size_t)BB * NN * CZ;          // [B,N,CZ] f32
  _Float16* hbase = (_Float16*)(rightv + (size_t)BB * NN * CZ);
  _Float16* Wh_epos = hbase;                 // 128*32
  _Float16* Wl_epos = Wh_epos + CZ * POS;
  _Float16* Wh_e1   = Wl_epos + CZ * POS;    // 128*128
  _Float16* Wl_e1   = Wh_e1 + CZ * CZ;
  _Float16* Wh_e2   = Wl_e1 + CZ * CZ;
  _Float16* Wl_e2   = Wh_e2 + CZ * CZ;

  // ---- weight conversion to f16 hi/lo ----
  cvt_kernel<<<(CZ * POS + 255) / 256, 256, 0, stream>>>(W_epos, Wh_epos,
                                                         Wl_epos, CZ * POS);
  cvt_kernel<<<(CZ * CZ + 255) / 256, 256, 0, stream>>>(W_e1, Wh_e1, Wl_e1,
                                                        CZ * CZ);
  cvt_kernel<<<(CZ * CZ + 255) / 256, 256, 0, stream>>>(W_e2, Wh_e2, Wl_e2,
                                                        CZ * CZ);

  node_kernel<<<dim3(NN, BB, 1), 256, 0, stream>>>(
      seq_idx, seq_feat, timestep, frame_mask,
      W_npos, b_npos, W_ntf, b_ntf, W_ntime, b_ntime,
      W_n1, b_n1, W_n2, b_n2, g_n, be_n,
      W_etf, W_etime, out_node, leftv, rightv);

  edge_kernel<<<dim3(NN / 4, NN / 16, BB), 128, 0, stream>>>(
      seq_idx, Wh_epos, Wl_epos, b_epos, b_etf, b_etime,
      Wh_e1, Wl_e1, b_e1, Wh_e2, Wl_e2, b_e2, g_e, be_e,
      leftv, rightv, out_edge);
}
